// ParallelLlamaAttention_47785806135393
// MI455X (gfx1250) — matrix operand order repaired and checked by reading, not compiled, not run on hardware
//
#include <hip/hip_runtime.h>
#include <hip/hip_bf16.h>

// ---------------------------------------------------------------------------
// CDNA5 (gfx1250) wave32 WMMA implementation of Llama attention block.
// Round 3: fix async-builtin pointer types (int4 in as(1)/as(3)).
// ---------------------------------------------------------------------------

typedef __attribute__((ext_vector_type(16))) __bf16 v16bf;
typedef __attribute__((ext_vector_type(8)))  float  v8f;
typedef __attribute__((ext_vector_type(4)))  int    v4i;

union FragBF {
  v16bf v;
  unsigned int u[8];
  unsigned short us[16];
};

// ---- CDNA5 async global->LDS path (bypasses VGPRs, tracked by ASYNCcnt) ----
#if defined(__has_builtin)
#if __has_builtin(__builtin_amdgcn_global_load_async_to_lds_b128)
#define HAVE_ASYNC_LDS 1
#endif
#endif

#ifdef HAVE_ASYNC_LDS
typedef __attribute__((address_space(1))) v4i global_v4i;
typedef __attribute__((address_space(3))) v4i lds_v4i;
__device__ __forceinline__ void async_copy_b128(const unsigned short* g,
                                                unsigned short* l) {
  __builtin_amdgcn_global_load_async_to_lds_b128(
      (global_v4i*)g, (lds_v4i*)l, 0, 0);
}
__device__ __forceinline__ void wait_async0() {
#if __has_builtin(__builtin_amdgcn_s_wait_asynccnt)
  __builtin_amdgcn_s_wait_asynccnt(0);
#else
  asm volatile("s_wait_asynccnt 0x0" ::: "memory");
#endif
}
#endif

__device__ __forceinline__ unsigned short f32_to_bf16(float f) {
  unsigned int u = __float_as_uint(f);
  u += 0x7FFFu + ((u >> 16) & 1u);   // round-to-nearest-even
  return (unsigned short)(u >> 16);
}

__device__ __forceinline__ v8f v8f_zero() {
  v8f z;
#pragma unroll
  for (int i = 0; i < 8; ++i) z[i] = 0.0f;
  return z;
}

// Reduce across the 16 lanes of a half-wave (xor masks 1,2,4,8 stay in-half).
__device__ __forceinline__ float half_reduce_max(float v) {
#pragma unroll
  for (int m = 1; m < 16; m <<= 1) v = fmaxf(v, __shfl_xor(v, m, 32));
  return v;
}
__device__ __forceinline__ float half_reduce_add(float v) {
#pragma unroll
  for (int m = 1; m < 16; m <<= 1) v += __shfl_xor(v, m, 32);
  return v;
}

// ---------------------------------------------------------------------------
// Kernel: fp32 -> bf16 conversion (grid-stride).
// ---------------------------------------------------------------------------
__global__ __launch_bounds__(256) void cvt_f32_bf16_kernel(
    const float* __restrict__ in, unsigned short* __restrict__ out, size_t n) {
  size_t i = (size_t)blockIdx.x * 256 + threadIdx.x;
  size_t stride = (size_t)gridDim.x * 256;
  for (; i < n; i += stride) out[i] = f32_to_bf16(in[i]);
}

// ---------------------------------------------------------------------------
// Kernel: C[M,N] (f32) = A[M,K] (bf16, row-major) @ W[N,K]^T (bf16, row-major)
// Block tile 128x128, BK=32, 256 threads = 8 waves (2x4 wave grid),
// each wave computes 64x32 via 4x2 WMMA 16x16x32 bf16 tiles.
// Double-buffered LDS; tile t+1 staged (async when available) while tile t
// feeds the matrix pipes.
// ---------------------------------------------------------------------------
__global__ __launch_bounds__(256) void gemm_bf16_wmma_kernel(
    const unsigned short* __restrict__ A,
    const unsigned short* __restrict__ W,
    float* __restrict__ C, int M, int N, int K) {
  constexpr int LDT = 40;  // padded LDS stride (elements)
  __shared__ __align__(16) unsigned short As[2][128 * LDT];
  __shared__ __align__(16) unsigned short Bs[2][128 * LDT];

  const int tid  = threadIdx.x;
  const int lane = tid & 31;
  const int wave = tid >> 5;
  const int l15  = lane & 15;
  const int hh   = lane >> 4;     // half-wave: 0 or 1
  const int wm   = wave >> 2;     // 0..1 -> 64-row block
  const int wn   = wave & 3;      // 0..3 -> 32-col block

  const int m0 = blockIdx.y * 128;
  const int n0 = blockIdx.x * 128;

  v8f acc[4][2];
#pragma unroll
  for (int i = 0; i < 4; ++i)
#pragma unroll
    for (int j = 0; j < 2; ++j) acc[i][j] = v8f_zero();

  const int ldrow = tid >> 2;        // 0..63 (+64 on 2nd pass)
  const int ldcol = (tid & 3) * 8;   // 0,8,16,24
  const int NT = K >> 5;             // number of 32-wide K tiles

  auto issue_tile = [&](int buf, int kt) {
#pragma unroll
    for (int p = 0; p < 2; ++p) {
      int row = ldrow + p * 64;
      const unsigned short* ga = A + (size_t)(m0 + row) * K + kt + ldcol;
      const unsigned short* gb = W + (size_t)(n0 + row) * K + kt + ldcol;
      unsigned short* la = &As[buf][row * LDT + ldcol];
      unsigned short* lb = &Bs[buf][row * LDT + ldcol];
#ifdef HAVE_ASYNC_LDS
      async_copy_b128(ga, la);
      async_copy_b128(gb, lb);
#else
      *(uint4*)la = *(const uint4*)ga;
      *(uint4*)lb = *(const uint4*)gb;
#endif
    }
  };

  issue_tile(0, 0);

  for (int t = 0; t < NT; ++t) {
    const int cur = t & 1;
#ifdef HAVE_ASYNC_LDS
    wait_async0();                 // tile t's async writes have landed in LDS
#endif
    __syncthreads();               // visible to all waves; prev buffer free
    if (t + 1 < NT) issue_tile(cur ^ 1, (t + 1) * 32);

    const unsigned short* Ac = As[cur];
    const unsigned short* Bc = Bs[cur];

    FragBF a[4], b[2];
#pragma unroll
    for (int mb = 0; mb < 4; ++mb) {
      int row = wm * 64 + mb * 16 + l15;
#pragma unroll
      for (int j = 0; j < 8; ++j) {
        // A 16x32 layout (matches B): half 0 -> K 0..15, half 1 -> K 16..31
        int k = hh * 16 + 2 * j;
        a[mb].u[j] = *(const unsigned int*)(Ac + row * LDT + k);
      }
    }
#pragma unroll
    for (int nb = 0; nb < 2; ++nb) {
      int row = wn * 32 + nb * 16 + l15;
#pragma unroll
      for (int j = 0; j < 8; ++j) {
        // B 32x16 layout (matches A): half 0 -> K 0..15, half 1 -> K 16..31
        int k = hh * 16 + 2 * j;
        b[nb].u[j] = *(const unsigned int*)(Bc + row * LDT + k);
      }
    }
#pragma unroll
    for (int mb = 0; mb < 4; ++mb)
#pragma unroll
      for (int nb = 0; nb < 2; ++nb)
        acc[mb][nb] = __builtin_amdgcn_wmma_f32_16x16x32_bf16(
            false, a[mb].v, false, b[nb].v, (short)0, acc[mb][nb], false, false);
  }

#pragma unroll
  for (int mb = 0; mb < 4; ++mb)
#pragma unroll
    for (int nb = 0; nb < 2; ++nb)
#pragma unroll
      for (int j = 0; j < 8; ++j) {
        int row = m0 + wm * 64 + mb * 16 + hh * 8 + j;  // C layout: M = 8*half + j
        int col = n0 + wn * 32 + nb * 16 + l15;         // N = lane%16
        C[(size_t)row * N + col] = acc[mb][nb][j];
      }
}

// ---------------------------------------------------------------------------
// Kernel: RoPE + QKV head split.  qkv f32 [4096,6144] ->
//   Q bf16 [B,32,S,128] (RoPE), K bf16 [B,8,S,128] (RoPE), V bf16 [B,8,S,128]
// ---------------------------------------------------------------------------
__global__ __launch_bounds__(256) void rope_split_kernel(
    const float* __restrict__ qkv, const int* __restrict__ pos_ids,
    unsigned short* __restrict__ Qg, unsigned short* __restrict__ Kg,
    unsigned short* __restrict__ Vg) {
  const int t = blockIdx.x;          // token 0..4095
  const int b = t >> 11;
  const int s = t & 2047;
  const float pos = (float)pos_ids[t];
  const float* row = qkv + (size_t)t * 6144;

  for (int n = threadIdx.x; n < 6144; n += 256) {
    float x = row[n];
    if (n < 4096) {                       // Q
      int hq = n >> 7, d = n & 127;
      float inv = __powf(10000.0f, -(float)(2 * (d & 63)) * (1.0f / 128.0f));
      float sn, cs;
      __sincosf(pos * inv, &sn, &cs);
      float partner = row[n ^ 64];        // d<64 <-> d+64 within head
      float rot = (d < 64) ? -partner : partner;
      Qg[(((size_t)(b * 32 + hq)) * 2048 + s) * 128 + d] = f32_to_bf16(x * cs + rot * sn);
    } else if (n < 5120) {                // K
      int m = n - 4096, hk = m >> 7, d = m & 127;
      float inv = __powf(10000.0f, -(float)(2 * (d & 63)) * (1.0f / 128.0f));
      float sn, cs;
      __sincosf(pos * inv, &sn, &cs);
      float partner = row[n ^ 64];
      float rot = (d < 64) ? -partner : partner;
      Kg[(((size_t)(b * 8 + hk)) * 2048 + s) * 128 + d] = f32_to_bf16(x * cs + rot * sn);
    } else {                              // V (no RoPE)
      int m = n - 5120, hv = m >> 7, d = m & 127;
      Vg[(((size_t)(b * 8 + hv)) * 2048 + s) * 128 + d] = f32_to_bf16(x);
    }
  }
}

// ---------------------------------------------------------------------------
// Kernel: causal flash attention, GQA 32q/8kv heads, D=128, S=2048.
// One workgroup (8 waves) per (b, h, 128 q rows); each wave owns 16 q rows.
// QK^T and PV via WMMA bf16; online softmax in registers (C-layout rows map
// to VGPR slots, so row stats are per-slot scalars after 16-lane reductions).
// K tiles staged via async global->LDS; V transposed manually.
// ---------------------------------------------------------------------------
__global__ __launch_bounds__(256) void flash_attn_kernel(
    const unsigned short* __restrict__ Qg, const unsigned short* __restrict__ Kg,
    const unsigned short* __restrict__ Vg, unsigned short* __restrict__ Og) {
  __shared__ __align__(16) unsigned short Ks[32 * 136];   // [key][dim], padded
  __shared__ __align__(16) unsigned short Vs[128 * 40];   // [dim][key], padded
  __shared__ __align__(16) unsigned short Pw[8 * 16 * 40];// per-wave P staging

  const int bid = blockIdx.x;               // b*512 + h*16 + qt
  const int qt  = bid & 15;
  const int h   = (bid >> 4) & 31;
  const int b   = bid >> 9;
  const int hkv = h >> 2;

  const int tid  = threadIdx.x;
  const int lane = tid & 31;
  const int wave = tid >> 5;
  const int l15  = lane & 15;
  const int hh   = lane >> 4;

  const unsigned short* Qbase = Qg + ((size_t)(b * 32 + h)) * 2048 * 128;
  const unsigned short* Kbase = Kg + ((size_t)(b * 8 + hkv)) * 2048 * 128;
  const unsigned short* Vbase = Vg + ((size_t)(b * 8 + hkv)) * 2048 * 128;

  const int q0 = qt * 128 + wave * 16;      // wave's first q row

  // Q fragments (16 rows x 128 dims = 4 A-fragments), kept in registers.
  FragBF qf[4];
  {
    const int qrow = q0 + l15;
#pragma unroll
    for (int c = 0; c < 4; ++c)
#pragma unroll
      for (int j = 0; j < 8; ++j) {
        int k = c * 32 + hh * 16 + 2 * j;
        qf[c].u[j] = *(const unsigned int*)(Qbase + (size_t)qrow * 128 + k);
      }
  }

  v8f o[8];
#pragma unroll
  for (int f = 0; f < 8; ++f) o[f] = v8f_zero();
  float mrun[8], lrun[8];
#pragma unroll
  for (int j = 0; j < 8; ++j) { mrun[j] = -1e30f; lrun[j] = 0.0f; }

  const float scale = 0.08838834764831845f;  // 1/sqrt(128)
  const int nkt = qt * 4 + 4;                // causal: keys up to qt*128+127

  for (int kt = 0; kt < nkt; ++kt) {
    const int key0 = kt * 32;
    __syncthreads();   // previous iteration's LDS reads complete
#pragma unroll
    for (int p = 0; p < 2; ++p) {
      int c = tid + p * 256;                 // 0..511 chunks of 8 bf16
      int r  = c >> 4;                       // key row 0..31
      int d0 = (c & 15) * 8;                 // dim start
      const unsigned short* kp = Kbase + (size_t)(key0 + r) * 128 + d0;
#ifdef HAVE_ASYNC_LDS
      async_copy_b128(kp, &Ks[r * 136 + d0]);
#else
      *(uint4*)(Ks + r * 136 + d0) = *(const uint4*)kp;
#endif
      const unsigned short* vp = Vbase + (size_t)(key0 + r) * 128 + d0;
#pragma unroll
      for (int i = 0; i < 8; ++i) Vs[(d0 + i) * 40 + r] = vp[i];  // transpose
    }
#ifdef HAVE_ASYNC_LDS
    wait_async0();
#endif
    __syncthreads();

    // ---- scores: S = Q (16x128) x K^T (128x32) -> two 16x16 fragments ----
    v8f sc[2];
#pragma unroll
    for (int nb = 0; nb < 2; ++nb) {
      sc[nb] = v8f_zero();
      const int keyr = nb * 16 + l15;
#pragma unroll
      for (int c = 0; c < 4; ++c) {
        FragBF kb;
#pragma unroll
        for (int j = 0; j < 8; ++j) {
          int k = c * 32 + hh * 16 + 2 * j;  // B-fragment: n=key lane, k=dim
          kb.u[j] = *(const unsigned int*)(Ks + keyr * 136 + k);
        }
        sc[nb] = __builtin_amdgcn_wmma_f32_16x16x32_bf16(
            false, qf[c].v, false, kb.v, (short)0, sc[nb], false, false);
      }
    }

    // ---- online softmax (row m = 8*hh + j lives in VGPR slot j) ----
    float p0[8], p1[8], rmax[8];
#pragma unroll
    for (int j = 0; j < 8; ++j) {
      int q = q0 + hh * 8 + j;
      float v0 = sc[0][j] * scale;
      float v1 = sc[1][j] * scale;
      if (key0 + l15 > q)      v0 = -1e30f;
      if (key0 + 16 + l15 > q) v1 = -1e30f;
      p0[j] = v0; p1[j] = v1;
      rmax[j] = half_reduce_max(fmaxf(v0, v1));
    }
#pragma unroll
    for (int j = 0; j < 8; ++j) {
      float mnew = fmaxf(mrun[j], rmax[j]);
      float corr = __expf(mrun[j] - mnew);
      mrun[j] = mnew;
      float e0 = __expf(p0[j] - mnew);
      float e1 = __expf(p1[j] - mnew);
      lrun[j] = lrun[j] * corr + half_reduce_add(e0 + e1);
#pragma unroll
      for (int f = 0; f < 8; ++f) o[f][j] = o[f][j] * corr;
      int prow = hh * 8 + j;
      Pw[wave * 640 + prow * 40 + l15]      = f32_to_bf16(e0);
      Pw[wave * 640 + prow * 40 + 16 + l15] = f32_to_bf16(e1);
    }

    // ---- PV: O += P (16x32) x V (32x128); P re-staged via per-wave LDS ----
    FragBF pa;
#pragma unroll
    for (int j = 0; j < 8; ++j) {
      int k = hh * 16 + 2 * j;
      pa.u[j] = *(const unsigned int*)(Pw + wave * 640 + l15 * 40 + k);
    }
#pragma unroll
    for (int f = 0; f < 8; ++f) {
      FragBF vb;
      const int d = f * 16 + l15;
#pragma unroll
      for (int j = 0; j < 8; ++j) {
        int k = hh * 16 + 2 * j;             // keys contiguous in Vs[dim][key]
        vb.u[j] = *(const unsigned int*)(Vs + d * 40 + k);
      }
      o[f] = __builtin_amdgcn_wmma_f32_16x16x32_bf16(
          false, pa.v, false, vb.v, (short)0, o[f], false, false);
    }
  }

  // ---- finalize: O /= l, write attn bf16 [B,S,H*D] ----
#pragma unroll
  for (int f = 0; f < 8; ++f) {
    const int col = h * 128 + f * 16 + l15;
#pragma unroll
    for (int j = 0; j < 8; ++j) {
      int q = q0 + hh * 8 + j;
      size_t t = (size_t)b * 2048 + q;
      Og[t * 4096 + col] = f32_to_bf16(o[f][j] / lrun[j]);
    }
  }
}

// ---------------------------------------------------------------------------
// Host-side launcher.
// ---------------------------------------------------------------------------
extern "C" void kernel_launch(void* const* d_in, const int* in_sizes, int n_in,
                              void* d_out, int out_size, void* d_ws, size_t ws_size,
                              hipStream_t stream) {
  (void)in_sizes; (void)n_in; (void)out_size; (void)ws_size;

  const float* hidden = (const float*)d_in[0];   // [2,2048,4096] f32
  const int*   posids = (const int*)d_in[1];     // [2,2048]
  const float* w_qkv  = (const float*)d_in[2];   // [6144,4096] f32
  const float* w_o    = (const float*)d_in[3];   // [4096,4096] f32
  float* out = (float*)d_out;                    // [2,2048,4096] f32

  const size_t M = 4096;        // tokens
  const size_t Nq = 6144, K = 4096, Nh = 4096;

  // Workspace carve-up (bf16 stored as unsigned short).
  unsigned short* hid_bf  = (unsigned short*)d_ws;          // 4096*4096
  unsigned short* wqkv_bf = hid_bf + M * K;                 // 6144*4096
  unsigned short* wo_bf   = wqkv_bf + Nq * K;               // 4096*4096
  float*          qkv_f   = (float*)(wo_bf + Nh * K);       // 4096*6144 f32
  unsigned short* Qb      = (unsigned short*)(qkv_f + M * Nq); // 2*32*2048*128
  unsigned short* Kb      = Qb + (size_t)2 * 32 * 2048 * 128;  // 2*8*2048*128
  unsigned short* Vb      = Kb + (size_t)2 * 8 * 2048 * 128;
  unsigned short* attn_bf = hid_bf;  // reuse: hidden_bf dead after QKV GEMM

  // 1) fp32 -> bf16 converts
  cvt_f32_bf16_kernel<<<4096, 256, 0, stream>>>(hidden, hid_bf, M * K);
  cvt_f32_bf16_kernel<<<4096, 256, 0, stream>>>(w_qkv, wqkv_bf, Nq * K);
  cvt_f32_bf16_kernel<<<4096, 256, 0, stream>>>(w_o, wo_bf, Nh * K);

  // 2) QKV GEMM: [4096,4096] x [6144,4096]^T -> [4096,6144] f32
  gemm_bf16_wmma_kernel<<<dim3(Nq / 128, M / 128), 256, 0, stream>>>(
      hid_bf, wqkv_bf, qkv_f, (int)M, (int)Nq, (int)K);

  // 3) RoPE + split into Q/K/V bf16 head-major layouts
  rope_split_kernel<<<4096, 256, 0, stream>>>(qkv_f, posids, Qb, Kb, Vb);

  // 4) Flash attention: grid = B * Hq * (S/128) = 2*32*16
  flash_attn_kernel<<<1024, 256, 0, stream>>>(Qb, Kb, Vb, attn_bf);

  // 5) Output projection: [4096,4096] x [4096,4096]^T -> d_out f32
  gemm_bf16_wmma_kernel<<<dim3(Nh / 128, M / 128), 256, 0, stream>>>(
      attn_bf, wo_bf, out, (int)M, (int)Nh, (int)K);
}
